// BrainwaveAttentionLayer_61392262529464
// MI455X (gfx1250) — matrix operand order repaired and checked by reading, not compiled, not run on hardware
//
#include <hip/hip_runtime.h>
#include <math.h>

// ---------------- types / constants ----------------
typedef __bf16 bf16;
typedef __attribute__((ext_vector_type(16))) __bf16 v16bf;
typedef __attribute__((ext_vector_type(8)))  __bf16 v8bf;
typedef __attribute__((ext_vector_type(4)))  float  v4f;
typedef __attribute__((ext_vector_type(8)))  float  v8f;

#define NB 16
#define NS 2048
#define ND 512
#define NS2 2052                             // NS + 4 halo rows (2 each side)
static const long BSD = 16L * 2048 * 512;    // elements per [B,S,D] tensor
static const long FXN = 16L * NS2 * 512;     // padded fx tensor elements

__device__ __forceinline__ v8f wmma_bf16(v16bf a, v16bf b, v8f c) {
  return __builtin_amdgcn_wmma_f32_16x16x32_bf16(false, a, false, b, (short)0, c, false, false);
}
__device__ __forceinline__ float clip10(float v) { return fminf(fmaxf(v, -10.f), 10.f); }
__device__ __forceinline__ bf16  tob(float f)    { return (bf16)f; }
__device__ __forceinline__ float fromb(bf16 h)   { return (float)h; }

// A fragment (16x32 bf16): lane l holds row m0+(l&15); element e -> K offset:
__device__ __forceinline__ int a_kof(int half, int e) {
  return (e < 8) ? (8 * half + e) : (8 * half + e + 8);
}

// CDNA5 async global->LDS copies (ASYNCcnt) -------------------------------
__device__ __forceinline__ void async_b32(unsigned ldsoff, const void* g) {
  asm volatile("global_load_async_to_lds_b32 %0, %1, off" :: "v"(ldsoff), "v"(g) : "memory");
}
__device__ __forceinline__ void async_b64(unsigned ldsoff, const void* g) {
  asm volatile("global_load_async_to_lds_b64 %0, %1, off" :: "v"(ldsoff), "v"(g) : "memory");
}
__device__ __forceinline__ void async_b128(unsigned ldsoff, const void* g) {
  asm volatile("global_load_async_to_lds_b128 %0, %1, off" :: "v"(ldsoff), "v"(g) : "memory");
}
__device__ __forceinline__ void wait_async0() {
  asm volatile("s_wait_asynccnt 0" ::: "memory");
}
__device__ __forceinline__ unsigned lds_off(const void* p) {
  return (unsigned)(size_t)p;                // LDS aperture: addr[31:0] == LDS byte offset
}

// load two contiguous 8xbf16 runs into a 16-element A fragment
__device__ __forceinline__ v16bf a_frag_bf16(const bf16* c0p, const bf16* c1p) {
  v8bf a0 = *(const v8bf*)c0p, a1 = *(const v8bf*)c1p;
  v16bf av;
#pragma unroll
  for (int e = 0; e < 8; ++e) { av[e] = a0[e]; av[8 + e] = a1[e]; }
  return av;
}
// same for f32 source (4x b128 + pack cvt)
__device__ __forceinline__ v16bf a_frag_f32(const float* c0p, const float* c1p) {
  v4f f0 = *(const v4f*)c0p, f1 = *(const v4f*)(c0p + 4);
  v4f f2 = *(const v4f*)c1p, f3 = *(const v4f*)(c1p + 4);
  v16bf av;
#pragma unroll
  for (int e = 0; e < 4; ++e) {
    av[e] = tob(f0[e]); av[4 + e] = tob(f1[e]);
    av[8 + e] = tob(f2[e]); av[12 + e] = tob(f3[e]);
  }
  return av;
}

// ---------------- small prep kernels ----------------
__global__ void k_fnorm(const float* __restrict__ ff, float* __restrict__ fn) {
  int br = blockIdx.x;
  __shared__ float red[256];
  float s = 0.f;
  for (int d = threadIdx.x; d < ND; d += 256) { float v = ff[br * ND + d]; s += v * v; }
  red[threadIdx.x] = s; __syncthreads();
  for (int o = 128; o; o >>= 1) { if (threadIdx.x < o) red[threadIdx.x] += red[threadIdx.x + o]; __syncthreads(); }
  float inv = 1.f / fmaxf(sqrtf(red[0]), 1e-12f);
  for (int d = threadIdx.x; d < ND; d += 256) fn[br * ND + d] = ff[br * ND + d] * inv;
}

// h'[br][t] = irfft(scaling)[t] - c0[br]   (DC removed so bf16 keeps structure)
__global__ void k_hfilter(bf16* __restrict__ hb,
                          float c00, float c01, float c02, float c03, float c04) {
  int idx = blockIdx.x * blockDim.x + threadIdx.x;     // 5*2048
  if (idx >= 5 * NS) return;
  int br = idx >> 11, t = idx & (NS - 1);
  const float betas[5] = {1.5f, 0.75f, 0.5f, 0.3f, 0.1f};
  const float c0s[5]   = {c00, c01, c02, c03, c04};
  float beta = betas[br];
  float acc = 0.f;
  for (int k = 0; k <= NS / 2; ++k) {
    float f = (float)k / (float)NS;
    float den = powf(f + 1e-8f, beta);
    den = fminf(fmaxf(den, 1e-8f), 1e6f);
    float sc = 1.f / den;
    float w = (k == 0 || k == NS / 2) ? 1.f : 2.f;
    int ph = (k * t) & (NS - 1);
    acc += w * sc * cosf(6.2831853071795864f * (float)ph / (float)NS);
  }
  hb[idx] = tob(acc / (float)NS - c0s[br]);
}

// zero the 2-row halos of fx
__global__ void k_zero_halo(bf16* __restrict__ fxb) {
  int idx = blockIdx.x * blockDim.x + threadIdx.x;     // 16*4*512
  if (idx >= NB * 4 * ND) return;
  int b = idx >> 11, r = (idx >> 9) & 3, d = idx & (ND - 1);
  int row = (r < 2) ? r : (NS + r);                    // rows 0,1,2050,2051
  fxb[((long)b * NS2 + row) * ND + d] = tob(0.f);
}

// fx = clip(x * fn, +-10)  -> bf16 into padded layout
__global__ void k_fx(const float* __restrict__ x, const float* __restrict__ fn,
                     bf16* __restrict__ fxb, int br) {
  long i = (long)blockIdx.x * blockDim.x + threadIdx.x;
  if (i >= BSD) return;
  int d = (int)(i & (ND - 1));
  int s = (int)((i >> 9) & (NS - 1));
  int b = (int)(i >> 20);
  fxb[((long)b * NS2 + s + 2) * ND + d] = tob(clip10(x[i] * fn[br * ND + d]));
}

// ---------------- grouped conv (k=5, groups=8) as WMMA GEMM + activation ----------------
__global__ __launch_bounds__(256) void k_conv_act(
    const float* __restrict__ x, const bf16* __restrict__ fxb,
    const float* __restrict__ cw, const float* __restrict__ cb,
    const float* __restrict__ fn, bf16* __restrict__ actb, int br) {
  __shared__ float panelf[32 * 16];                    // B panel (f32), 2KB
  int tid = threadIdx.x, wave = tid >> 5, lane = tid & 31;
  int nt = blockIdx.x & 31, mt2 = (blockIdx.x >> 5) & 15, b = blockIdx.x >> 9;
  int s0 = mt2 * 128 + wave * 16, o0 = nt * 16, g = o0 >> 6;
  int mrow = lane & 15, half = lane >> 4, ncol = lane & 15, kb = 16 * half;
  const bf16* arow = fxb + ((long)b * NS2 + (s0 + mrow)) * ND + g * 64;
  v8f acc = {};
  for (int k0 = 0; k0 < 320; k0 += 32) {               // K = 5 taps * 64 in-chan
#pragma unroll
    for (int j = 0; j < 2; ++j) {                      // stage gathered weights
      int idx = tid * 2 + j, kk = idx >> 4, n = idx & 15;
      int k = k0 + kk, tap = k >> 6, ci = k & 63;
      async_b32(lds_off(&panelf[kk * 16 + n]),
                (const void*)&cw[(((long)br * ND + (o0 + n)) * 64 + ci) * 5 + tap]);
    }
    wait_async0();
    __syncthreads();
    // A: two contiguous 8-elem runs (each run stays inside one 64-wide tap block)
    int c0 = k0 + 8 * half, c1 = k0 + 16 + 8 * half;
    v16bf av = a_frag_bf16(arow + (long)(c0 >> 6) * ND + (c0 & 63),
                           arow + (long)(c1 >> 6) * ND + (c1 & 63));
    v16bf bv;
#pragma unroll
    for (int e = 0; e < 16; ++e) bv[e] = tob(panelf[a_kof(half, e) * 16 + ncol]);
    acc = wmma_bf16(av, bv, acc);
    __syncthreads();
  }
#pragma unroll
  for (int r = 0; r < 8; ++r) {
    int s = s0 + r + 8 * half, o = o0 + ncol;
    float conv = acc[r] + cb[br * ND + o];
    float a;
    switch (br) {
      case 0:  a = 0.5f * conv * (1.f + erff(conv * 0.70710678118f)); break;
      case 1:  a = fmaxf(conv, 0.f); break;
      case 2:  a = tanhf(conv); break;
      case 3:  a = 1.f / (1.f + expf(-conv)); break;
      default: a = log1pf(expf(-fabsf(conv))) + fmaxf(conv, 0.f); break;
    }
    float fx = clip10(x[((long)b * NS + s) * ND + o] * fn[br * ND + o]);
    actb[((long)b * NS + s) * ND + o] = tob(a + 0.1f * fx);
  }
}

// column sums of activated (split-out DC rank-1 term)
__global__ void k_colsum(const bf16* __restrict__ actb, float* __restrict__ cs) {
  int idx = blockIdx.x * blockDim.x + threadIdx.x;
  if (idx >= NB * ND) return;
  int b = idx >> 9, d = idx & (ND - 1);
  float s = 0.f;
  const bf16* p = actb + (long)b * NS * ND + d;
  for (int t = 0; t < NS; ++t) s += fromb(p[(long)t * ND]);
  cs[idx] = s;
}

// noise[s,d] = sum_t h'[(s-t)&2047]*act[t,d] + c0*colsum[d]; 2-wide N tiles per wave
__global__ __launch_bounds__(256) void k_spectral(
    const bf16* __restrict__ actb, const bf16* __restrict__ hbr,
    const float* __restrict__ cs, float* __restrict__ noise, float c0) {
  __shared__ bf16 hsh[NS];                             // full h' (4KB) resident
  __shared__ bf16 panel[32 * 32];                      // B panel (bf16), 2KB
  int tid = threadIdx.x, wave = tid >> 5, lane = tid & 31;
  int nt = blockIdx.x & 15, mt2 = (blockIdx.x >> 4) & 15, b = blockIdx.x >> 8;
  int d0 = nt * 32, s0 = mt2 * 128 + wave * 16;
  int mrow = lane & 15, half = lane >> 4, ncol = lane & 15, kb = 16 * half;

  async_b128(lds_off(&hsh[tid * 8]), (const void*)(hbr + tid * 8));  // 16B/thread
  wait_async0();
  __syncthreads();

  v8f acc0 = {}, acc1 = {};
  const bf16* bbase = actb + (long)b * NS * ND + d0;
  for (int k0 = 0; k0 < NS; k0 += 32) {
    {
      int kk = tid >> 3, np = (tid & 7) * 4;           // 4 consecutive bf16 = b64
      async_b64(lds_off(&panel[kk * 32 + np]),
                (const void*)(bbase + (long)(k0 + kk) * ND + np));
    }
    wait_async0();
    __syncthreads();
    v16bf av, bv0, bv1;
#pragma unroll
    for (int e = 0; e < 16; ++e) {
      int ka = k0 + kb + e;
      av[e]  = hsh[((s0 + mrow) - ka) & (NS - 1)];
      bv0[e] = panel[(kb + e) * 32 + ncol];
      bv1[e] = panel[(kb + e) * 32 + ncol + 16];
    }
    acc0 = wmma_bf16(av, bv0, acc0);
    acc1 = wmma_bf16(av, bv1, acc1);
    __syncthreads();
  }
#pragma unroll
  for (int r = 0; r < 8; ++r) {
    int s = s0 + r + 8 * half;
    int da = d0 + ncol, db = d0 + ncol + 16;
    long rowb = ((long)b * NS + s) * ND;
    noise[rowb + da] = clip10(acc0[r] + c0 * cs[b * ND + da]);
    noise[rowb + db] = clip10(acc1[r] + c0 * cs[b * ND + db]);
  }
}

// in-place row LayerNorm over last dim (512)
__global__ void k_ln(float* __restrict__ buf, const float* __restrict__ w,
                     const float* __restrict__ bb) {
  long row = blockIdx.x;
  float* p = buf + row * ND;
  __shared__ float red[256];
  float s = 0.f;
  for (int d = threadIdx.x; d < ND; d += 256) s += p[d];
  red[threadIdx.x] = s; __syncthreads();
  for (int o = 128; o; o >>= 1) { if (threadIdx.x < o) red[threadIdx.x] += red[threadIdx.x + o]; __syncthreads(); }
  float mu = red[0] / (float)ND; __syncthreads();
  float s2 = 0.f;
  for (int d = threadIdx.x; d < ND; d += 256) { float t = p[d] - mu; s2 += t * t; }
  red[threadIdx.x] = s2; __syncthreads();
  for (int o = 128; o; o >>= 1) { if (threadIdx.x < o) red[threadIdx.x] += red[threadIdx.x + o]; __syncthreads(); }
  float inv = rsqrtf(red[0] / (float)ND + 1e-5f); __syncthreads();
  for (int d = threadIdx.x; d < ND; d += 256) p[d] = (p[d] - mu) * inv * w[d] + bb[d];
}

// EMA scan along S
__global__ void k_ema(const float* __restrict__ noise, bf16* __restrict__ outb, int window) {
  int idx = blockIdx.x * blockDim.x + threadIdx.x;
  if (idx >= NB * ND) return;
  int b = idx >> 9, d = idx & (ND - 1);
  float a = 1.f / (float)window;
  const float* p = noise + (long)b * NS * ND + d;
  bf16* q = outb + (long)b * NS * ND + d;
  float m = p[0]; q[0] = tob(m);
  for (int s = 1; s < NS; ++s) {
    m = a * p[(long)s * ND] + (1.f - a) * m;
    q[(long)s * ND] = tob(m);
  }
}

// proj_pre = concat(outs) @ det_w + det_b   (K = 2560)
__global__ __launch_bounds__(256) void k_det(
    const bf16* __restrict__ outs, const float* __restrict__ dw,
    const float* __restrict__ db, float* __restrict__ proj) {
  __shared__ float panelf[32 * 16];
  int tid = threadIdx.x, wave = tid >> 5, lane = tid & 31;
  int nt = blockIdx.x & 31, mt2 = (blockIdx.x >> 5) & 15, b = blockIdx.x >> 9;
  int s0 = mt2 * 128 + wave * 16, n0 = nt * 16;
  int mrow = lane & 15, half = lane >> 4, ncol = lane & 15, kb = 16 * half;
  long rowb = ((long)b * NS + s0 + mrow) * ND;
  v8f acc = {};
  for (int k0 = 0; k0 < 5 * ND; k0 += 32) {
    {
      int kk = tid >> 3, np = (tid & 7) * 2;           // 2 consecutive f32 = b64
      async_b64(lds_off(&panelf[kk * 16 + np]),
                (const void*)&dw[(long)(k0 + kk) * ND + n0 + np]);
    }
    wait_async0();
    __syncthreads();
    int c0 = k0 + 8 * half, c1 = k0 + 16 + 8 * half;   // runs stay in one branch slice
    v16bf av = a_frag_bf16(outs + (long)(c0 >> 9) * BSD + rowb + (c0 & (ND - 1)),
                           outs + (long)(c1 >> 9) * BSD + rowb + (c1 & (ND - 1)));
    v16bf bv;
#pragma unroll
    for (int e = 0; e < 16; ++e) bv[e] = tob(panelf[a_kof(half, e) * 16 + ncol]);
    acc = wmma_bf16(av, bv, acc);
    __syncthreads();
  }
#pragma unroll
  for (int r = 0; r < 8; ++r) {
    int s = s0 + r + 8 * half, n = n0 + ncol;
    proj[((long)b * NS + s) * ND + n] = acc[r] + db[n];
  }
}

// coherence[b,s] = sigmoid( sum_{i<j} |coupling[i,j]| * cos_sim(out_i,out_j) / 10 )
__global__ void k_coherence(const bf16* __restrict__ outs, const float* __restrict__ cpl,
                            float* __restrict__ coh) {
  long row = blockIdx.x, base = row * ND;
  int tid = threadIdx.x;
  float nn[5] = {0, 0, 0, 0, 0}, dot[10] = {0, 0, 0, 0, 0, 0, 0, 0, 0, 0};
  for (int d = tid; d < ND; d += 256) {
    float v[5];
#pragma unroll
    for (int i = 0; i < 5; ++i) v[i] = fromb(outs[(long)i * BSD + base + d]);
    int p = 0;
#pragma unroll
    for (int i = 0; i < 5; ++i) {
      nn[i] += v[i] * v[i];
#pragma unroll
      for (int j = i + 1; j < 5; ++j) dot[p++] += v[i] * v[j];
    }
  }
  __shared__ float red[15][256];
#pragma unroll
  for (int i = 0; i < 5; ++i) red[i][tid] = nn[i];
#pragma unroll
  for (int p = 0; p < 10; ++p) red[5 + p][tid] = dot[p];
  __syncthreads();
  for (int o = 128; o; o >>= 1) {
    if (tid < o)
      for (int q = 0; q < 15; ++q) red[q][tid] += red[q][tid + o];
    __syncthreads();
  }
  if (tid == 0) {
    float nrm[5];
#pragma unroll
    for (int i = 0; i < 5; ++i) nrm[i] = fmaxf(sqrtf(red[i][0]), 1e-8f);
    float s = 0.f; int p = 0;
    for (int i = 0; i < 5; ++i)
      for (int j = i + 1; j < 5; ++j) { s += fabsf(cpl[i * 5 + j]) * (red[5 + p][0] / (nrm[i] * nrm[j])); ++p; }
    coh[row] = 1.f / (1.f + expf(-s * 0.1f));
  }
}

// exact k-th order statistic via binary search on float bit patterns (values in (0,1))
__device__ unsigned kth_bits(const float* coh, int rank, int tid, int* red) {
  unsigned lo = 0u, hi = 0x3F800000u;
  while (lo < hi) {
    unsigned mid = lo + ((hi - lo) >> 1);
    int c = 0;
    for (int i = tid; i < NB * NS; i += 1024) c += (__float_as_uint(coh[i]) <= mid) ? 1 : 0;
    red[tid] = c; __syncthreads();
    for (int o = 512; o; o >>= 1) { if (tid < o) red[tid] += red[tid + o]; __syncthreads(); }
    int total = red[0]; __syncthreads();
    if (total >= rank + 1) hi = mid; else lo = mid + 1;
  }
  return lo;
}

__global__ __launch_bounds__(1024) void k_quantile(const float* __restrict__ coh,
                                                   float* __restrict__ thr) {
  __shared__ int red[1024];
  int tid = threadIdx.x;
  const int n = NB * NS;
  float pos = 0.7f * (float)(n - 1);
  int k = (int)floorf(pos);
  unsigned b0 = kth_bits(coh, k, tid, red);
  unsigned b1 = kth_bits(coh, k + 1, tid, red);
  if (tid == 0) {
    float v0 = __uint_as_float(b0), v1 = __uint_as_float(b1);
    float q = v0 + (pos - (float)k) * (v1 - v0);
    *thr = fminf(fmaxf(q, 0.1f), 0.9f);
  }
}

// punched = proj*(1-hole*0.5) + (proj@punch_w+punch_b)*c*1.5 + cstate*c*0.1, clipped
__global__ __launch_bounds__(256) void k_final(
    const float* __restrict__ proj, const float* __restrict__ pw,
    const float* __restrict__ pb, const float* __restrict__ cst,
    const float* __restrict__ coh, const float* __restrict__ thrp,
    float* __restrict__ out) {
  __shared__ float panelf[32 * 16];
  int tid = threadIdx.x, wave = tid >> 5, lane = tid & 31;
  int nt = blockIdx.x & 31, mt2 = (blockIdx.x >> 5) & 15, b = blockIdx.x >> 9;
  int s0 = mt2 * 128 + wave * 16, n0 = nt * 16;
  int mrow = lane & 15, half = lane >> 4, ncol = lane & 15, kb = 16 * half;
  const float* arow = proj + ((long)b * NS + s0 + mrow) * ND;
  v8f acc = {};
  for (int k0 = 0; k0 < ND; k0 += 32) {
    {
      int kk = tid >> 3, np = (tid & 7) * 2;
      async_b64(lds_off(&panelf[kk * 16 + np]),
                (const void*)&pw[(long)(k0 + kk) * ND + n0 + np]);
    }
    wait_async0();
    __syncthreads();
    v16bf av = a_frag_f32(arow + k0 + 8 * half, arow + k0 + 16 + 8 * half);
    v16bf bv;
#pragma unroll
    for (int e = 0; e < 16; ++e) bv[e] = tob(panelf[a_kof(half, e) * 16 + ncol]);
    acc = wmma_bf16(av, bv, acc);
    __syncthreads();
  }
  float thr = *thrp;
#pragma unroll
  for (int r = 0; r < 8; ++r) {
    int s = s0 + r + 8 * half, n = n0 + ncol;
    float c = coh[(long)b * NS + s];
    float hole = (c > thr) ? 0.5f : 0.f;
    float p = proj[((long)b * NS + s) * ND + n];
    float e = acc[r] + pb[n];
    float v = p * (1.f - hole) + e * c * 1.5f + cst[n] * c * 0.1f;
    out[((long)b * NS + s) * ND + n] = clip10(v);
  }
}

// ---------------- host launch ----------------
static float host_c0(double beta) {
  double den = pow(1e-8, beta);
  if (den < 1e-8) den = 1e-8;
  if (den > 1e6)  den = 1e6;
  return (float)(1.0 / den / (double)NS);
}

extern "C" void kernel_launch(void* const* d_in, const int* in_sizes, int n_in,
                              void* d_out, int out_size, void* d_ws, size_t ws_size,
                              hipStream_t stream) {
  const float* x      = (const float*)d_in[0];
  const float* ff     = (const float*)d_in[1];
  const float* conv_w = (const float*)d_in[2];
  const float* conv_b = (const float*)d_in[3];
  const float* ln_w   = (const float*)d_in[4];
  const float* ln_b   = (const float*)d_in[5];
  const float* det_w  = (const float*)d_in[6];
  const float* det_b  = (const float*)d_in[7];
  const float* cpl    = (const float*)d_in[8];
  const float* pw     = (const float*)d_in[9];
  const float* pb     = (const float*)d_in[10];
  const float* cst    = (const float*)d_in[11];
  const float* slw    = (const float*)d_in[12];
  const float* slb    = (const float*)d_in[13];
  float* out = (float*)d_out;

  char* ws = (char*)d_ws;
  float* fn   = (float*)(ws + 0);            // 5*512 f32
  bf16*  hb   = (bf16*)(ws + 64 * 1024);     // 5*2048 bf16
  float* cs   = (float*)(ws + 128 * 1024);   // 16*512 f32
  float* coh  = (float*)(ws + 256 * 1024);   // 16*2048 f32
  float* thr  = (float*)(ws + 512 * 1024);   // 1 f32
  size_t off = 1u << 20;
  bf16*  fxb  = (bf16*)(ws + off);  off += (size_t)FXN * 2;     // padded fx
  bf16*  actb = (bf16*)(ws + off);  off += (size_t)BSD * 2;
  float* noise= (float*)(ws + off); off += (size_t)BSD * 4;
  bf16*  outs = (bf16*)(ws + off);  off += (size_t)BSD * 2 * 5;
  float* proj = (float*)(ws + off); off += (size_t)BSD * 4;
  (void)ws_size; (void)in_sizes; (void)n_in; (void)out_size;

  const double betas[5] = {1.5, 0.75, 0.5, 0.3, 0.1};
  const int windows[5]  = {4, 8, 16, 32, 64};
  float c0[5];
  for (int i = 0; i < 5; ++i) c0[i] = host_c0(betas[i]);

  k_fnorm<<<5, 256, 0, stream>>>(ff, fn);
  k_hfilter<<<(5 * NS + 255) / 256, 256, 0, stream>>>(hb, c0[0], c0[1], c0[2], c0[3], c0[4]);

  const int gemm_blocks = 8192;       // (b=16) x (mt2=16) x (nt=32), 16-wide N
  const int gemm_blocks2 = 4096;      // (b=16) x (mt2=16) x (nt=16), 32-wide N
  for (int br = 0; br < 5; ++br) {
    k_zero_halo<<<(NB * 4 * ND) / 256, 256, 0, stream>>>(fxb);
    k_fx<<<(int)(BSD / 256), 256, 0, stream>>>(x, fn, fxb, br);
    k_conv_act<<<gemm_blocks, 256, 0, stream>>>(x, fxb, conv_w, conv_b, fn, actb, br);
    k_colsum<<<(NB * ND) / 256, 256, 0, stream>>>(actb, cs);
    k_spectral<<<gemm_blocks2, 256, 0, stream>>>(actb, hb + (long)br * NS, cs, noise, c0[br]);
    k_ln<<<NB * NS, 256, 0, stream>>>(noise, ln_w + br * ND, ln_b + br * ND);
    k_ema<<<(NB * ND) / 256, 256, 0, stream>>>(noise, outs + (long)br * BSD, windows[br]);
  }

  k_det<<<gemm_blocks, 256, 0, stream>>>(outs, det_w, det_b, proj);
  k_ln<<<NB * NS, 256, 0, stream>>>(proj, slw, slb);
  k_coherence<<<NB * NS, 256, 0, stream>>>(outs, cpl, coh);
  k_quantile<<<1, 1024, 0, stream>>>(coh, thr);
  k_final<<<gemm_blocks, 256, 0, stream>>>(proj, pw, pb, cst, coh, thr, out);
}
